// TemporalSpikeGraphConvNet_14396730376457
// MI455X (gfx1250) — matrix operand order repaired and checked by reading, not compiled, not run on hardware
//
#include <hip/hip_runtime.h>
#include <hip/hip_bf16.h>

// ---------------- model hyperparameters (compile-time) ----------------
#define B_    4
#define T_    112
#define N_    10000
#define FIN_  7
#define H_    32
#define HOR_  24
#define K_    2
#define E_    160000
#define BN_   (B_ * N_)          // 40000 rows
#define ALPHA_ 0.9f
#define BETA_  0.8f
#define THR_   1.0f

// GEMM1: feats[BN,160] @ diff_w[160,32]     (K blocks of 32 -> 5)
// GEMM2: z[BN,32]      @ dec_w[32,168]      (pad N to 176 -> 11 tiles of 16)
#define K1_    ((2 * K_ + 1) * H_)   // 160
#define N2_    (FIN_ * HOR_)         // 168
#define N2PAD_ 176
#define NT2_   (N2PAD_ / 16)         // 11

typedef __attribute__((ext_vector_type(16))) _Float16 v16h;
typedef __attribute__((ext_vector_type(8)))  float    v8f;

// K position of element e (of a v16h frag) for half-wave g, contiguous order
// (lanes 0-15: K 0..15, lanes 16-31: K 16..31), matching the B staging.
__device__ __forceinline__ int a_kloc(int e, int g) {
    return 16 * g + e;
}

// ---------------------------------------------------------------------
// 0) zero scratch (atomic accumulation targets + degrees)
// ---------------------------------------------------------------------
__global__ void zero_f32_kernel(float* __restrict__ p, int n) {
    int i = blockIdx.x * blockDim.x + threadIdx.x;
    if (i < n) p[i] = 0.0f;
}

// ---------------------------------------------------------------------
// 1) fused encode + spiking recurrence. One wave per (b,n); lane = h.
//    spk_out layout: [BN, H] row-major (row = b*N + n).
// ---------------------------------------------------------------------
__global__ __launch_bounds__(256)
void encode_scan_kernel(const float* __restrict__ x,
                        const float* __restrict__ enc_w,   // [FIN, H]
                        const float* __restrict__ enc_b,   // [H]
                        const float* __restrict__ node_emb,// [N, H]
                        float* __restrict__ spk_out)       // [BN, H]
{
    const int wave = threadIdx.x >> 5;
    const int lane = threadIdx.x & 31;          // = h channel
    int p = blockIdx.x * 8 + wave;              // (b,n) pair index
    if (p >= BN_) return;
    p = __builtin_amdgcn_readfirstlane(p);      // wave-uniform -> scalar addr math
    const int b = p / N_;
    const int n = p - b * N_;

    // per-lane encoder column + time-invariant bias
    float wf[FIN_];
#pragma unroll
    for (int f = 0; f < FIN_; ++f) wf[f] = enc_w[f * H_ + lane];
    const float base = enc_b[lane] + node_emb[n * H_ + lane];

    float syn = 0.0f, mem = 0.0f, spk = 0.0f;
    const size_t row_stride = (size_t)N_ * FIN_;          // stride over t
    const float* xp = x + ((size_t)b * T_ * N_ + n) * FIN_;

    for (int t = 0; t < T_; ++t) {
        if (t + 8 < T_)
            __builtin_prefetch(xp + (size_t)(t + 8) * row_stride, 0, 3);
        const float* xr = xp + (size_t)t * row_stride;    // wave-uniform address
        float cur = base;
#pragma unroll
        for (int f = 0; f < FIN_; ++f) cur = fmaf(xr[f], wf[f], cur);
        syn = ALPHA_ * syn + cur;
        mem = BETA_ * mem + syn - spk * THR_;
        spk = (mem > THR_) ? 1.0f : 0.0f;
    }
    spk_out[(size_t)p * H_ + lane] = spk;
}

// ---------------------------------------------------------------------
// 2) degree accumulation: deg_f[dst] += w, deg_b[src] += w
// ---------------------------------------------------------------------
__global__ void degree_kernel(const int* __restrict__ src,
                              const int* __restrict__ dst,
                              const float* __restrict__ w,
                              float* __restrict__ deg_f,
                              float* __restrict__ deg_b)
{
    int e = blockIdx.x * blockDim.x + threadIdx.x;
    if (e >= E_) return;
    float we = w[e];
    atomicAdd(&deg_f[dst[e]], we);
    atomicAdd(&deg_b[src[e]], we);
}

// ---------------------------------------------------------------------
// 3) one weighted aggregation hop:
//    out[b, scat[e], h] += (w[e]/max(deg[scat[e]],1e-8)) * in[b, gath[e], h]
//    128 threads per edge (tid = b*32 + h).
// ---------------------------------------------------------------------
__global__ __launch_bounds__(256)
void agg_kernel(const float* __restrict__ in,   // [BN, H]
                float* __restrict__ out,        // [BN, H]
                const int* __restrict__ gath,
                const int* __restrict__ scat,
                const float* __restrict__ deg,
                const float* __restrict__ w)
{
    long long idx = (long long)blockIdx.x * blockDim.x + threadIdx.x;
    int e = (int)(idx >> 7);
    if (e >= E_) return;
    int r  = (int)(idx & 127);
    int b  = r >> 5;
    int h  = r & 31;
    int gn = gath[e];
    int sn = scat[e];
    float coeff = w[e] / fmaxf(deg[sn], 1e-8f);
    float val = in[((size_t)b * N_ + gn) * H_ + h] * coeff;
    atomicAdd(&out[((size_t)b * N_ + sn) * H_ + h], val);
}

// ---------------------------------------------------------------------
// 4) fused diffusion-matmul + decode + output transpose, via f16 WMMA
//    with f32 accumulation. One wave per 16-row M tile (2500 tiles).
//    Weights are staged into LDS in FRAGMENT-MAJOR f16 order so each
//    WMMA B operand is two ds_load_b128 instead of 16x ds_load_u16.
// ---------------------------------------------------------------------
__global__ __launch_bounds__(128)
void fused_gemm_kernel(const float* __restrict__ spk,
                       const float* __restrict__ hf1,
                       const float* __restrict__ hf2,
                       const float* __restrict__ hb1,
                       const float* __restrict__ hb2,
                       const float* __restrict__ diff_w, // [160,32]
                       const float* __restrict__ diff_b, // [32]
                       const float* __restrict__ dec_w,  // [32,168]
                       const float* __restrict__ dec_b,  // [168]
                       float* __restrict__ out)          // [B,HOR,N,FIN]
{
    // fragment-major B images: frag[(kblock*ntiles + nt)*32 + lane][16]
    __shared__ __attribute__((aligned(32))) _Float16 sF1[5 * 2 * 32 * 16];    // 10 KB
    __shared__ __attribute__((aligned(32))) _Float16 sF2[NT2_ * 32 * 16];     // 11 KB
    __shared__ float sZ[4][16 * 33];   // per-wave z tile, stride 33 (bank-safe)

    const int tid = threadIdx.x;

    // ---- stage diff_w into GEMM1 B fragments (B 32x16, contiguous k order: k = 16*(lane/16) + e) ----
    for (int i = tid; i < 5 * 2 * 32 * 16; i += 128) {
        int e    = i & 15;
        int ln   = (i >> 4) & 31;
        int rem  = i >> 9;          // kb*2 + nt
        int nt   = rem & 1;
        int kb   = rem >> 1;
        int k    = kb * 32 + 16 * (ln >> 4) + e;
        int n    = nt * 16 + (ln & 15);
        sF1[i] = (_Float16)diff_w[k * H_ + n];
    }
    // ---- stage dec_w into GEMM2 B fragments (N padded to 176) ----
    for (int i = tid; i < NT2_ * 32 * 16; i += 128) {
        int e  = i & 15;
        int ln = (i >> 4) & 31;
        int t2 = i >> 9;
        int k  = 16 * (ln >> 4) + e;
        int n  = t2 * 16 + (ln & 15);
        sF2[i] = (n < N2_) ? (_Float16)dec_w[k * N2_ + n] : (_Float16)0.0f;
    }
    __syncthreads();

    const int wave  = tid >> 5;
    const int lane  = tid & 31;
    const int g     = lane >> 4;     // half-wave
    const int col   = lane & 15;
    const int mtile = blockIdx.x * 4 + wave;   // < 2500 exactly

    const float* chunks[5] = { spk, hf1, hf2, hb1, hb2 };

    // ---------------- GEMM1: z[16,32] = feats_tile[16,160] @ diff_w ----------------
    v8f c0 = {}; v8f c1 = {};
    const size_t rowA = (size_t)(mtile * 16 + col) * H_;  // A-frag row = lane%16
#pragma unroll
    for (int kb = 0; kb < 5; ++kb) {          // K blocks of 32 == feats chunks
        const float* ch = chunks[kb] + rowA;
        v16h a;
#pragma unroll
        for (int e = 0; e < 16; ++e) a[e] = (_Float16)ch[a_kloc(e, g)];
        const v16h b0 = *(const v16h*)&sF1[((kb * 2 + 0) * 32 + lane) * 16];
        const v16h b1 = *(const v16h*)&sF1[((kb * 2 + 1) * 32 + lane) * 16];
        c0 = __builtin_amdgcn_wmma_f32_16x16x32_f16(false, a, false, b0,
                                                    (short)0, c0, false, false);
        c1 = __builtin_amdgcn_wmma_f32_16x16x32_f16(false, a, false, b1,
                                                    (short)0, c1, false, false);
    }

    // stage z (+diff_b) into LDS to re-layout C(16x16 f32) -> A(16x32 f16)
    const float db0 = diff_b[col];
    const float db1 = diff_b[16 + col];
#pragma unroll
    for (int r = 0; r < 8; ++r) {
        int m = r + 8 * g;                    // ISA C layout: lanes16-31 hold M=8..15
        sZ[wave][m * 33 + col]      = c0[r] + db0;
        sZ[wave][m * 33 + 16 + col] = c1[r] + db1;
    }
    __syncthreads();

    v16h a2;
#pragma unroll
    for (int e = 0; e < 16; ++e)
        a2[e] = (_Float16)sZ[wave][col * 33 + a_kloc(e, g)];

    // ---------------- GEMM2: out_tile[16,168] = z[16,32] @ dec_w ----------------
    for (int t2 = 0; t2 < NT2_; ++t2) {
        const v16h b = *(const v16h*)&sF2[(t2 * 32 + lane) * 16];
        v8f zacc = {};
        v8f d = __builtin_amdgcn_wmma_f32_16x16x32_f16(false, a2, false, b,
                                                       (short)0, zacc, false, false);
#pragma unroll
        for (int r = 0; r < 8; ++r) {
            int m    = r + 8 * g;
            int grow = mtile * 16 + m;        // global row in [0, BN)
            int cgl  = t2 * 16 + col;         // global col in [0, 176)
            if (cgl < N2_) {
                float val = d[r] + dec_b[cgl];
                int bb   = grow / N_;
                int node = grow - bb * N_;
                int tt   = cgl / FIN_;
                int ff   = cgl - tt * FIN_;
                // out[b, t, n, f]
                out[(((size_t)bb * HOR_ + tt) * N_ + node) * FIN_ + ff] = val;
            }
        }
    }
}

// ---------------------------------------------------------------------
// host-side launch
// ---------------------------------------------------------------------
extern "C" void kernel_launch(void* const* d_in, const int* in_sizes, int n_in,
                              void* d_out, int out_size, void* d_ws, size_t ws_size,
                              hipStream_t stream) {
    const float* x        = (const float*)d_in[0];
    const int*   ei       = (const int*)  d_in[1];   // [2, E]: src then dst
    const float* ew       = (const float*)d_in[2];
    const float* enc_w    = (const float*)d_in[3];
    const float* enc_b    = (const float*)d_in[4];
    const float* node_emb = (const float*)d_in[5];
    const float* diff_w   = (const float*)d_in[6];
    const float* diff_b   = (const float*)d_in[7];
    const float* dec_w    = (const float*)d_in[8];
    const float* dec_b    = (const float*)d_in[9];
    float*       out      = (float*)d_out;

    const int* src = ei;
    const int* dst = ei + E_;

    // workspace layout (floats): spk | hf1 | hf2 | hb1 | hb2 | deg_f | deg_b
    float* spk   = (float*)d_ws;
    float* hf1   = spk + (size_t)BN_ * H_;
    float* hf2   = hf1 + (size_t)BN_ * H_;
    float* hb1   = hf2 + (size_t)BN_ * H_;
    float* hb2   = hb1 + (size_t)BN_ * H_;
    float* deg_f = hb2 + (size_t)BN_ * H_;
    float* deg_b = deg_f + N_;

    // 0) zero atomic targets (hf1..hb2 + degrees are contiguous)
    {
        int nz = 4 * BN_ * H_ + 2 * N_;
        zero_f32_kernel<<<(nz + 255) / 256, 256, 0, stream>>>(hf1, nz);
    }

    // 1) encode + spiking scan -> spk  (8 waves / block, 1 wave per (b,n))
    encode_scan_kernel<<<(BN_ + 7) / 8, 256, 0, stream>>>(x, enc_w, enc_b,
                                                          node_emb, spk);

    // 2) degrees
    degree_kernel<<<(E_ + 255) / 256, 256, 0, stream>>>(src, dst, ew, deg_f, deg_b);

    // 3) diffusion hops (forward: gather src -> scatter dst, norm by deg_f[dst];
    //    backward: gather dst -> scatter src, norm by deg_b[src])
    const long long agg_threads = (long long)E_ * 128;
    const int agg_blocks = (int)((agg_threads + 255) / 256);
    agg_kernel<<<agg_blocks, 256, 0, stream>>>(spk, hf1, src, dst, deg_f, ew);
    agg_kernel<<<agg_blocks, 256, 0, stream>>>(hf1, hf2, src, dst, deg_f, ew);
    agg_kernel<<<agg_blocks, 256, 0, stream>>>(spk, hb1, dst, src, deg_b, ew);
    agg_kernel<<<agg_blocks, 256, 0, stream>>>(hb1, hb2, dst, src, deg_b, ew);

    // 4) WMMA diffusion-matmul + decode + transpose  (2500 M tiles / 4 per block)
    fused_gemm_kernel<<<(BN_ / 16) / 4, 128, 0, stream>>>(
        spk, hf1, hf2, hb1, hb2, diff_w, diff_b, dec_w, dec_b, out);
}
